// GCN_52201032516108
// MI455X (gfx1250) — matrix operand order repaired and checked by reading, not compiled, not run on hardware
//
#include <hip/hip_runtime.h>

// ---------------------------------------------------------------------------
// Problem constants (from the reference)
// ---------------------------------------------------------------------------
#define N_NODES 120832
#define N_EDGES 300000
#define HID     128
#define KST     3     // ARMA stacks
#define TIT     2     // ARMA iterations
#define NBUS    118
#define BATCHSZ 1024
#define NIN     8
#define NOUTF   4

typedef __attribute__((ext_vector_type(16))) __bf16 v16bf;
typedef __attribute__((ext_vector_type(8)))  float  v8f;

static inline int cdiv(int a, int b) { return (a + b - 1) / b; }

// ---------------------------------------------------------------------------
// Utility kernels
// ---------------------------------------------------------------------------
__global__ void zero_f32(float* __restrict__ p, size_t n) {
  size_t i = (size_t)blockIdx.x * blockDim.x + threadIdx.x;
  size_t stride = (size_t)gridDim.x * blockDim.x;
  for (; i < n; i += stride) p[i] = 0.0f;
}

__global__ void deg_scatter(const long long* __restrict__ cols,
                            const float* __restrict__ eattr,
                            float* __restrict__ deg, int E) {
  int e = blockIdx.x * blockDim.x + threadIdx.x;
  if (e < E) atomicAdd(&deg[cols[e]], fabsf(eattr[2 * e]));
}

__global__ void norm_compute(const long long* __restrict__ rows,
                             const long long* __restrict__ cols,
                             const float* __restrict__ eattr,
                             const float* __restrict__ deg,
                             float* __restrict__ nrm, int E) {
  int e = blockIdx.x * blockDim.x + threadIdx.x;
  if (e >= E) return;
  float dr = deg[rows[e]];
  float dc = deg[cols[e]];
  float ir = dr > 0.0f ? rsqrtf(dr) : 0.0f;
  float ic = dc > 0.0f ? rsqrtf(dc) : 0.0f;
  nrm[e] = ir * fabsf(eattr[2 * e]) * ic;
}

// D^-1/2 A D^-1/2 propagate, fused over the K=3 stacks.
// One thread handles one (edge, 4-feature group): float4 gather, scale, 4 atomics.
__global__ void propagate3(const float* __restrict__ src, float* __restrict__ dst,
                           const long long* __restrict__ rows,
                           const long long* __restrict__ cols,
                           const float* __restrict__ nrm,
                           int F, size_t nf, int E) {
  int gpe = F >> 2;
  size_t total = (size_t)E * gpe;
  size_t idx = (size_t)blockIdx.x * blockDim.x + threadIdx.x;
  if (idx >= total) return;
  int e = (int)(idx / gpe);
  int g = (int)(idx % gpe);
  long long r = rows[e], c = cols[e];
  float w = nrm[e];
#pragma unroll
  for (int k = 0; k < KST; ++k) {
    const float4 v = *(const float4*)(src + (size_t)k * nf + (size_t)r * F + g * 4);
    float* d = dst + (size_t)k * nf + (size_t)c * F + g * 4;
    atomicAdd(d + 0, v.x * w);
    atomicAdd(d + 1, v.y * w);
    atomicAdd(d + 2, v.z * w);
    atomicAdd(d + 3, v.w * w);
  }
}

// data = relu(data + root + bias[t][k][f])   over [K, N, F] (in place)
__global__ void ew_post(float* __restrict__ data, const float* __restrict__ root,
                        const float* __restrict__ bias, int F, size_t nf) {
  size_t idx = (size_t)blockIdx.x * blockDim.x + threadIdx.x;
  size_t total = (size_t)KST * nf;
  if (idx >= total) return;
  int k = (int)(idx / nf);
  int f = (int)(idx % (size_t)F);
  float v = data[idx] + root[idx] + bias[k * F + f];
  data[idx] = v > 0.0f ? v : 0.0f;
}

// y = mean over K stacks (+ optional outer relu)
__global__ void mean3(const float* __restrict__ b, size_t nf,
                      float* __restrict__ y, int do_relu) {
  size_t i = (size_t)blockIdx.x * blockDim.x + threadIdx.x;
  if (i >= nf) return;
  float v = (b[i] + b[i + nf] + b[i + 2 * nf]) * (1.0f / 3.0f);
  if (do_relu) v = fmaxf(v, 0.0f);
  y[i] = v;
}

// out[n][0:128] = h[n], out[n][128:136] = x[n]
__global__ void concat_hx(const float* __restrict__ h, const float* __restrict__ x,
                          float* __restrict__ out, int n) {
  int i = blockIdx.x * blockDim.x + threadIdx.x;
  int FT = HID + NIN;
  if (i >= n * FT) return;
  int node = i / FT;
  int f = i % FT;
  out[i] = (f < HID) ? h[(size_t)node * HID + f] : x[(size_t)node * NIN + (f - HID)];
}

// ---------------------------------------------------------------------------
// Weight packing into the CDNA5 WMMA bf16 B-fragment layout.
// Wp index = ((kstep*ntiles + ntile)*32 + lane)*16 + e
//   lane 0..15 : K = kstep*32 + (e&7) + (e>>3)*16,      n = ntile*16 + lane
//   lane 16..31: K = kstep*32 + 8 + (e&7) + (e>>3)*16,  n = ntile*16 + (lane-16)
// transpose=1 reads W as [Nout][Kdim] (for cls_w^T).
// ---------------------------------------------------------------------------
__global__ void pack_b(const float* __restrict__ W, int ldw, int Kdim, int Nout,
                       int transpose, __bf16* __restrict__ Wp,
                       int ksteps, int ntiles) {
  int i = blockIdx.x * blockDim.x + threadIdx.x;
  int total = ksteps * ntiles * 512;
  if (i >= total) return;
  int e = i & 15;
  int lane = (i >> 4) & 31;
  int tile = i >> 9;
  int ntile = tile % ntiles;
  int kstep = tile / ntiles;
  int k = kstep * 32 + (lane >> 4) * 8 + (e & 7) + (e >> 3) * 16;
  int n = ntile * 16 + (lane & 15);
  float v = 0.0f;
  if (k < Kdim && n < Nout)
    v = transpose ? W[(size_t)n * ldw + k] : W[(size_t)k * ldw + n];
  Wp[i] = (__bf16)v;
}

// ---------------------------------------------------------------------------
// WMMA GEMM: Y[M,Nout] = X[M,Kdim] @ W (+ optional bias), fp32 in/out,
// bf16 multiply via v_wmma_f32_16x16x32_bf16.
// One wave per 16-row strip; each wave carries NT=8 accumulators and covers
// 8 column tiles (grid.y covers ntiles beyond 8).
// Per K-step: A = four contiguous float4 loads + packed bf16 converts,
// B = 8 fragments staged into registers first (one clause of 16 b128 loads),
// then 8 back-to-back WMMAs. FULL=true removes all per-tile guards (used
// whenever ntiles % NT == 0, i.e. every Nout=128 GEMM).
//
// A fragment (16-bit A, 16x32): lane L row = L&15; half = L>>4:
//   a[e]   = X[row][ks*32 + half*8 + e]        e=0..7
//   a[8+e] = X[row][ks*32 + 16 + half*8 + e]
// D layout: lane L col = L&15, rows = half*8 + v (v = 0..7).
// ---------------------------------------------------------------------------
#define GEMM_NT 8

template <bool FULL>
__global__ void gemm_wmma_bf16(const float* __restrict__ X, int lda,
                               const __bf16* __restrict__ Wp,
                               float* __restrict__ Y, int ldy,
                               const float* __restrict__ bias,
                               int Mrows, int Kdim, int Nout,
                               int ksteps, int ntiles) {
  const int lane = threadIdx.x;
  const int mtile = blockIdx.x;
  const int tbase = blockIdx.y * GEMM_NT;  // first column tile of this wave
  const int hf = lane >> 4;
  const int mrow = mtile * 16 + (lane & 15);
  const float* xr = X + (size_t)(mrow < Mrows ? mrow : 0) * lda;
  // Base of this wave's B stream for kstep 0: fragments for tiles
  // tbase..tbase+7 are 512-element (1 KB) apart; next kstep is ntiles*512 on.
  const __bf16* wpl = Wp + ((size_t)tbase * 512 + (size_t)lane * 16);
  const size_t kstride = (size_t)ntiles * 512;

  v8f zero = {};
  v8f c[GEMM_NT];
#pragma unroll
  for (int t = 0; t < GEMM_NT; ++t) c[t] = zero;

  const int kfull = Kdim >> 5;
  for (int ks = 0; ks < kfull; ++ks) {
    // ---- A fragment: branch-free contiguous loads + converts
    const int kb = ks * 32 + hf * 8;
    const float4 f0 = *(const float4*)(xr + kb);
    const float4 f1 = *(const float4*)(xr + kb + 4);
    const float4 f2 = *(const float4*)(xr + kb + 16);
    const float4 f3 = *(const float4*)(xr + kb + 20);
    v16bf a;
    a[0]  = (__bf16)f0.x; a[1]  = (__bf16)f0.y; a[2]  = (__bf16)f0.z; a[3]  = (__bf16)f0.w;
    a[4]  = (__bf16)f1.x; a[5]  = (__bf16)f1.y; a[6]  = (__bf16)f1.z; a[7]  = (__bf16)f1.w;
    a[8]  = (__bf16)f2.x; a[9]  = (__bf16)f2.y; a[10] = (__bf16)f2.z; a[11] = (__bf16)f2.w;
    a[12] = (__bf16)f3.x; a[13] = (__bf16)f3.y; a[14] = (__bf16)f3.z; a[15] = (__bf16)f3.w;
    // ---- Stage all B fragments, then issue the WMMAs back-to-back.
    const __bf16* wk = wpl + (size_t)ks * kstride;
    v16bf bf[GEMM_NT];
#pragma unroll
    for (int t = 0; t < GEMM_NT; ++t)
      if (FULL || tbase + t < ntiles)
        bf[t] = *(const v16bf*)(wk + (size_t)t * 512);
#pragma unroll
    for (int t = 0; t < GEMM_NT; ++t)
      if (FULL || tbase + t < ntiles)
        c[t] = __builtin_amdgcn_wmma_f32_16x16x32_bf16(
            false, a, false, bf[t], (short)0, c[t], false, false);
  }
  if (kfull < ksteps) {  // guarded tail (Kdim % 32 != 0)
    const int kb = kfull * 32 + hf * 8;
    v16bf a;
#pragma unroll
    for (int e = 0; e < 8; ++e) {
      int k0 = kb + e;
      int k1 = kb + 16 + e;
      a[e]     = (k0 < Kdim) ? (__bf16)xr[k0] : (__bf16)0.0f;
      a[e + 8] = (k1 < Kdim) ? (__bf16)xr[k1] : (__bf16)0.0f;
    }
    const __bf16* wk = wpl + (size_t)kfull * kstride;
    v16bf bf[GEMM_NT];
#pragma unroll
    for (int t = 0; t < GEMM_NT; ++t)
      if (FULL || tbase + t < ntiles)
        bf[t] = *(const v16bf*)(wk + (size_t)t * 512);
#pragma unroll
    for (int t = 0; t < GEMM_NT; ++t)
      if (FULL || tbase + t < ntiles)
        c[t] = __builtin_amdgcn_wmma_f32_16x16x32_bf16(
            false, a, false, bf[t], (short)0, c[t], false, false);
  }

  // Store D tiles.
  const int m0 = mtile * 16 + hf * 8;
#pragma unroll
  for (int t = 0; t < GEMM_NT; ++t) {
    if (FULL || tbase + t < ntiles) {
      const int ncol = (tbase + t) * 16 + (lane & 15);
      if (ncol < Nout) {
        const float badd = bias ? bias[ncol] : 0.0f;
#pragma unroll
        for (int v = 0; v < 8; ++v) {
          int m = m0 + v;
          if (m < Mrows) Y[(size_t)m * ldy + ncol] = c[t][v] + badd;
        }
      }
    }
  }
}

// ---------------------------------------------------------------------------
// Host-side orchestration
// ---------------------------------------------------------------------------
struct ArmaParams {
  const float* bias;    // [T,K,Fout]
  const float* init_w;  // [K,Fin,Fout]
  const float* root_w;  // [T,K,Fin,Fout]
  const float* w;       // [T-1,K,Fout,Fout]
};

static void launch_pack(hipStream_t s, const float* W, int ldw, int Kd, int Nn,
                        int tr, __bf16* Wp) {
  int ksteps = cdiv(Kd, 32), ntiles = cdiv(Nn, 16);
  int total = ksteps * ntiles * 512;
  pack_b<<<cdiv(total, 256), 256, 0, s>>>(W, ldw, Kd, Nn, tr, Wp, ksteps, ntiles);
}

static void launch_gemm(hipStream_t s, const float* X, int lda, const __bf16* Wp,
                        float* Y, int ldy, const float* bias,
                        int M, int Kd, int Nn) {
  int ksteps = cdiv(Kd, 32), ntiles = cdiv(Nn, 16);
  dim3 g(cdiv(M, 16), cdiv(ntiles, GEMM_NT));
  if (ntiles % GEMM_NT == 0)
    gemm_wmma_bf16<true><<<g, 32, 0, s>>>(X, lda, Wp, Y, ldy, bias, M, Kd, Nn,
                                          ksteps, ntiles);
  else
    gemm_wmma_bf16<false><<<g, 32, 0, s>>>(X, lda, Wp, Y, ldy, bias, M, Kd, Nn,
                                           ksteps, ntiles);
}

static void launch_zero(hipStream_t s, float* p, size_t n) {
  zero_f32<<<4096, 256, 0, s>>>(p, n);
}

static void run_arma(hipStream_t s, const float* X, int Fin, int Fout,
                     const ArmaParams& p, float* Y, bool outer_relu,
                     float* B0, float* B1, float* B2, __bf16* wp,
                     const long long* rows, const long long* cols,
                     const float* nrm) {
  const int F = Fout;
  const size_t nf = (size_t)N_NODES * F;

  // state_k = X @ init_w[k]
  for (int k = 0; k < KST; ++k) {
    launch_pack(s, p.init_w + (size_t)k * Fin * F, F, Fin, F, 0, wp);
    launch_gemm(s, X, Fin, wp, B0 + (size_t)k * nf, F, nullptr, N_NODES, Fin, F);
  }

  float* state = B0;
  float* tmp = B1;
  for (int t = 0; t < TIT; ++t) {
    if (t > 0) {
      // state_k <- state_k @ w[t-1][k]
      for (int k = 0; k < KST; ++k) {
        launch_pack(s, p.w + (size_t)((t - 1) * KST + k) * F * F, F, F, F, 0, wp);
        launch_gemm(s, state + (size_t)k * nf, F, wp, tmp + (size_t)k * nf, F,
                    nullptr, N_NODES, F, F);
      }
      float* sw = state; state = tmp; tmp = sw;
    }
    // root_k = X @ root_w[t][k]  -> B2
    for (int k = 0; k < KST; ++k) {
      launch_pack(s, p.root_w + (size_t)(t * KST + k) * Fin * F, F, Fin, F, 0, wp);
      launch_gemm(s, X, Fin, wp, B2 + (size_t)k * nf, F, nullptr, N_NODES, Fin, F);
    }
    // propagate state -> tmp (zeroed), then tmp = relu(tmp + root + bias[t])
    launch_zero(s, tmp, (size_t)KST * nf);
    {
      size_t total = (size_t)N_EDGES * (F >> 2);
      propagate3<<<(unsigned)((total + 255) / 256), 256, 0, s>>>(
          state, tmp, rows, cols, nrm, F, nf, N_EDGES);
    }
    {
      size_t total = (size_t)KST * nf;
      ew_post<<<(unsigned)((total + 255) / 256), 256, 0, s>>>(
          tmp, B2, p.bias + (size_t)t * KST * F, F, nf);
    }
    float* sw = state; state = tmp; tmp = sw;
  }
  mean3<<<(unsigned)((nf + 255) / 256), 256, 0, s>>>(state, nf, Y, outer_relu ? 1 : 0);
}

extern "C" void kernel_launch(void* const* d_in, const int* in_sizes, int n_in,
                              void* d_out, int out_size, void* d_ws, size_t ws_size,
                              hipStream_t stream) {
  (void)in_sizes; (void)n_in; (void)out_size;

  // Input order: pytree flatten with sorted dict keys:
  // x, edge_index, edge_attr, [params: cls_b, cls_w,
  //   gnns0{bias,init_w,root_w,w}, gnns1{...}, loc0{...}, loc1{...},
  //   pinn0{...}, pinn1{...}], batch_size
  const float* x = (const float*)d_in[0];
  const long long* eidx = (const long long*)d_in[1];  // int64 [2,E]
  const float* eattr = (const float*)d_in[2];
  const float* cls_b = (const float*)d_in[3];
  const float* cls_w = (const float*)d_in[4];
  auto arma_at = [&](int base) {
    ArmaParams p;
    p.bias   = (const float*)d_in[base + 0];
    p.init_w = (const float*)d_in[base + 1];
    p.root_w = (const float*)d_in[base + 2];
    p.w      = (const float*)d_in[base + 3];
    return p;
  };
  ArmaParams gnn0 = arma_at(5),  gnn1 = arma_at(9);
  ArmaParams loc0 = arma_at(13), loc1 = arma_at(17);
  ArmaParams pin0 = arma_at(21), pin1 = arma_at(25);

  // Scratch layout (fp32 elements; all offsets stay 32B aligned).
  float* ws = (float*)d_ws;
  size_t o = 0;
  float* nrm   = ws + o; o += N_EDGES;
  float* deg   = ws + o; o += N_NODES;
  float* actA  = ws + o; o += (size_t)N_NODES * (HID + NIN);  // also cls input
  float* actB  = ws + o; o += (size_t)N_NODES * HID;
  float* hkeep = ws + o; o += (size_t)N_NODES * HID;
  float* B0 = ws + o; o += (size_t)KST * N_NODES * HID;
  float* B1 = ws + o; o += (size_t)KST * N_NODES * HID;
  float* B2 = ws + o; o += (size_t)KST * N_NODES * HID;
  __bf16* wp = (__bf16*)(ws + o);
  const size_t wp_bytes = (size_t)cdiv(NBUS * HID, 32) * cdiv(2 * NBUS, 16) * 512 * 2;
  if (ws_size < o * sizeof(float) + wp_bytes) return;  // scratch too small: no-op

  const long long* rows = eidx;
  const long long* cols = eidx + N_EDGES;

  // gcn_norm (no self loops)
  launch_zero(stream, deg, N_NODES);
  deg_scatter<<<cdiv(N_EDGES, 256), 256, 0, stream>>>(cols, eattr, deg, N_EDGES);
  norm_compute<<<cdiv(N_EDGES, 256), 256, 0, stream>>>(rows, cols, eattr, deg, nrm, N_EDGES);

  // gnns
  run_arma(stream, x,    NIN, HID, gnn0, actB,  true, B0, B1, B2, wp, rows, cols, nrm);
  run_arma(stream, actB, HID, HID, gnn1, hkeep, true, B0, B1, B2, wp, rows, cols, nrm);

  // pinn branch -> d_out[0 : N*4)
  run_arma(stream, hkeep, HID, HID,   pin0, actB, true,  B0, B1, B2, wp, rows, cols, nrm);
  run_arma(stream, actB,  HID, NOUTF, pin1, (float*)d_out, false, B0, B1, B2, wp, rows, cols, nrm);

  // loc branch: concat(h, x) -> 2 ARMA layers -> classifier
  concat_hx<<<cdiv(N_NODES * (HID + NIN), 256), 256, 0, stream>>>(hkeep, x, actA, N_NODES);
  run_arma(stream, actA, HID + NIN, HID, loc0, actB, true, B0, B1, B2, wp, rows, cols, nrm);
  run_arma(stream, actB, HID,       HID, loc1, actA, true, B0, B1, B2, wp, rows, cols, nrm);

  // loc.reshape(1024, 15104) @ cls_w.T + cls_b -> d_out[N*4 : N*4 + 1024*236)
  launch_pack(stream, cls_w, NBUS * HID, NBUS * HID, 2 * NBUS, 1, wp);
  launch_gemm(stream, actA, NBUS * HID, wp,
              (float*)d_out + (size_t)N_NODES * NOUTF, 2 * NBUS, cls_b,
              BATCHSZ, NBUS * HID, 2 * NBUS);
}
